// SchnetConv_71708773974042
// MI455X (gfx1250) — compile-verified, matrix operand order checked by reading, not run on hardware
//
#include <hip/hip_runtime.h>
#include <stdint.h>

// ---------------------------------------------------------------------------
// SchNet interaction block for gfx1250 (MI455X), bf16 WMMA path.
// N=25000 nodes, K=32 neighbors, R=256 rbf, H=128 hidden, F=64 features.
// ---------------------------------------------------------------------------

typedef __attribute__((ext_vector_type(16))) __bf16 v16bf;
typedef __attribute__((ext_vector_type(8)))  float  v8f;

union Frag16 {
    v16bf    bf;
    uint32_t u[8];
    uint4    q[2];
};

#define RDIM 256
#define H1   128
#define FDIM 64
#define KNEI 32
#define LN2F 0.6931471805599453f

// hot path: pack two f32 -> two bf16 (truncate) with a single v_perm_b32
__device__ __forceinline__ uint32_t pk_bf16(float lo, float hi) {
    return __builtin_amdgcn_perm(__builtin_bit_cast(uint32_t, hi),
                                 __builtin_bit_cast(uint32_t, lo),
                                 0x07060302u);
}

// hot path: single f32 -> bf16 (truncate), one shift
__device__ __forceinline__ uint16_t f2bf_fast(float x) {
    return (uint16_t)(__builtin_bit_cast(uint32_t, x) >> 16);
}

// cold path (weights): round-to-nearest-even
__device__ __forceinline__ uint16_t f2bf_rne(float x) {
    uint32_t u = __builtin_bit_cast(uint32_t, x);
    return (uint16_t)((u + 0x7FFFu + ((u >> 16) & 1u)) >> 16);
}

// shifted softplus: log(1+exp(x)) - ln2, numerically stable
__device__ __forceinline__ float sp_ln2(float x) {
    float ax = fabsf(x);
    return fmaxf(x, 0.0f) + __logf(1.0f + __expf(-ax)) - LN2F;
}

// ---------------------------------------------------------------------------
// Prep: convert W1[128x256], W2[64x128], W3[64x64], W4[64x64] f32 -> bf16 in ws
// ---------------------------------------------------------------------------
__global__ void cvt_weights_kernel(const float* __restrict__ W1,
                                   const float* __restrict__ W2,
                                   const float* __restrict__ W3,
                                   const float* __restrict__ W4,
                                   uint16_t* __restrict__ ws) {
    int i = blockIdx.x * 256 + threadIdx.x;
    if (i < 32768)      ws[i] = f2bf_rne(W1[i]);
    else if (i < 40960) ws[i] = f2bf_rne(W2[i - 32768]);
    else if (i < 45056) ws[i] = f2bf_rne(W3[i - 40960]);
    else if (i < 49152) ws[i] = f2bf_rne(W4[i - 45056]);
}

// ---------------------------------------------------------------------------
// Fused filter network + gather + K-reduction.
// One block = 4 nodes = 128 NK-rows. 8 waves; wave w owns rows [16w,16w+16).
// ---------------------------------------------------------------------------
__global__ __launch_bounds__(256) void schnet_filter_kernel(
    const float* __restrict__ bf,       // [N*K, 256]
    const float* __restrict__ hfeat,    // [N, 64]
    const float* __restrict__ h_edge,   // [N*K, 64]
    const float* __restrict__ cutoff,   // [N*K]
    const int*   __restrict__ knn,      // [N*K]
    const float* __restrict__ b1,       // [128]
    const float* __restrict__ b2,       // [64]
    const uint16_t* __restrict__ w1bf,  // [128][256] bf16
    const uint16_t* __restrict__ w2bf,  // [64][128]  bf16
    float* __restrict__ m_out,          // [N, 64]
    int N)
{
    constexpr int USTRIDE = H1 + 8;     // 136 elems; row stride 272B = 16B-mult
    __shared__ __attribute__((aligned(16))) uint16_t sW1[H1 * RDIM];     // 64KB
    __shared__ __attribute__((aligned(16))) uint16_t sW2[FDIM * H1];     // 16KB
    __shared__ __attribute__((aligned(16))) uint16_t sU[128 * USTRIDE];  // 34KB
    __shared__ __attribute__((aligned(16))) float    sM[4 * FDIM];       // 1KB

    const int tid   = threadIdx.x;
    const int wave  = tid >> 5;
    const int lane  = tid & 31;
    const int laneM = lane & 15;
    const int kgrp  = (lane >> 4) * 8;   // 0 or 8

    // stage weights into LDS
    {
        const uint32_t* s1 = (const uint32_t*)w1bf;
        uint32_t*       d1 = (uint32_t*)sW1;
        for (int i = tid; i < H1 * RDIM / 2; i += 256) d1[i] = s1[i];
        const uint32_t* s2 = (const uint32_t*)w2bf;
        uint32_t*       d2 = (uint32_t*)sW2;
        for (int i = tid; i < FDIM * H1 / 2; i += 256) d2[i] = s2[i];
    }
    for (int i = tid; i < 4 * FDIM; i += 256) sM[i] = 0.0f;
    __syncthreads();

    const long NK      = (long)N * KNEI;
    const long rowBase = (long)blockIdx.x * 128;
    long rowA = rowBase + wave * 16 + laneM;
    if (rowA >= NK) rowA = NK - 1;
    const float* aptr = bf + rowA * RDIM;
    // prefetch validity for the next block's rows (speculative anyway)
    const bool pfOK = (rowA + 128 < NK);

    const v8f vzero = {};

    // ---------------- GEMM1: [128,256] x [256,128] (bf16 WMMA) -------------
    v8f acc1[8];
#pragma unroll
    for (int t = 0; t < 8; ++t) acc1[t] = vzero;

    for (int ks = 0; ks < 8; ++ks) {
        const int k0 = ks * 32;
        // hint: next block's A rows stream next (L2 warm-up)
        if (pfOK) __builtin_prefetch(aptr + 128 * RDIM + k0 + kgrp, 0, 3);
        // A fragment: VGPR0-3 = K k0+kgrp+0..7, VGPR4-7 = K k0+16+kgrp+0..7
        const float4 f0 = *(const float4*)(aptr + k0 + kgrp);
        const float4 f1 = *(const float4*)(aptr + k0 + kgrp + 4);
        const float4 g0 = *(const float4*)(aptr + k0 + 16 + kgrp);
        const float4 g1 = *(const float4*)(aptr + k0 + 16 + kgrp + 4);
        Frag16 afrag;
        afrag.u[0] = pk_bf16(f0.x, f0.y);
        afrag.u[1] = pk_bf16(f0.z, f0.w);
        afrag.u[2] = pk_bf16(f1.x, f1.y);
        afrag.u[3] = pk_bf16(f1.z, f1.w);
        afrag.u[4] = pk_bf16(g0.x, g0.y);
        afrag.u[5] = pk_bf16(g0.z, g0.w);
        afrag.u[6] = pk_bf16(g1.x, g1.y);
        afrag.u[7] = pk_bf16(g1.z, g1.w);
#pragma unroll
        for (int t = 0; t < 8; ++t) {
            const int n = t * 16 + laneM;               // B column = W1 row
            const uint16_t* wrow = sW1 + n * RDIM + k0 + kgrp;
            Frag16 bfrag;
            bfrag.q[0] = *(const uint4*)(wrow);
            bfrag.q[1] = *(const uint4*)(wrow + 16);
            acc1[t] = __builtin_amdgcn_wmma_f32_16x16x32_bf16(
                false, afrag.bf, false, bfrag.bf, (short)0, acc1[t], false, false);
        }
    }

    // bias + softplus, store bf16 activation tile to LDS
#pragma unroll
    for (int t = 0; t < 8; ++t) {
        const int col   = t * 16 + laneM;
        const float bb  = b1[col];
#pragma unroll
        for (int v = 0; v < 8; ++v) {
            const int rloc = wave * 16 + v + kgrp;      // kgrp == 8*(lane>=16)
            sU[rloc * USTRIDE + col] = f2bf_fast(sp_ln2(acc1[t][v] + bb));
        }
    }
    __syncthreads();

    // ---------------- GEMM2: [128,128] x [128,64] ---------------------------
    v8f acc2[4];
#pragma unroll
    for (int t = 0; t < 4; ++t) acc2[t] = vzero;

    const uint16_t* urow = sU + (wave * 16 + laneM) * USTRIDE;
#pragma unroll
    for (int ks = 0; ks < 4; ++ks) {
        const int k0 = ks * 32;
        Frag16 afrag;
        afrag.q[0] = *(const uint4*)(urow + k0 + kgrp);
        afrag.q[1] = *(const uint4*)(urow + k0 + 16 + kgrp);
#pragma unroll
        for (int t = 0; t < 4; ++t) {
            const int n = t * 16 + laneM;
            const uint16_t* wrow = sW2 + n * H1 + k0 + kgrp;
            Frag16 bfrag;
            bfrag.q[0] = *(const uint4*)(wrow);
            bfrag.q[1] = *(const uint4*)(wrow + 16);
            acc2[t] = __builtin_amdgcn_wmma_f32_16x16x32_bf16(
                false, afrag.bf, false, bfrag.bf, (short)0, acc2[t], false, false);
        }
    }

    // ------ softplus -> w, scale by cutoff * h[knn] * h_edge, reduce over K -
    const int nodeLoc = wave >> 1;                      // 2 waves per node
    float psum[4] = {0.0f, 0.0f, 0.0f, 0.0f};
#pragma unroll
    for (int v = 0; v < 8; ++v) {
        const int  rloc = wave * 16 + v + kgrp;
        long rglb = rowBase + rloc;
        if (rglb >= NK) rglb = NK - 1;
        const float ct  = cutoff[rglb];
        const int   idx = knn[rglb];
        const float* hrow  = hfeat  + (long)idx * FDIM;
        const float* herow = h_edge + rglb * FDIM;
#pragma unroll
        for (int t = 0; t < 4; ++t) {
            const int col = t * 16 + laneM;
            const float wv = sp_ln2(acc2[t][v] + b2[col]);
            psum[t] += wv * ct * hrow[col] * herow[col];
        }
    }
#pragma unroll
    for (int t = 0; t < 4; ++t)
        atomicAdd(&sM[nodeLoc * FDIM + t * 16 + laneM], psum[t]);
    __syncthreads();

    for (int i = tid; i < 4 * FDIM; i += 256) {
        const long node = (long)blockIdx.x * 4 + (i >> 6);
        if (node < N) m_out[node * FDIM + (i & 63)] = sM[i];
    }
}

// ---------------------------------------------------------------------------
// Output MLPs: out = mlp(mlp(m, W3,b3), W4,b4). 128 nodes per block, 8 waves.
// ---------------------------------------------------------------------------
__global__ __launch_bounds__(256) void schnet_out_kernel(
    const float* __restrict__ m,        // [N,64]
    const float* __restrict__ b3,
    const float* __restrict__ b4,
    const uint16_t* __restrict__ w3bf,  // [64][64] bf16
    const uint16_t* __restrict__ w4bf,  // [64][64] bf16
    float* __restrict__ out,            // [N,64]
    int N)
{
    constexpr int TSTRIDE = FDIM + 16;  // 80 elems; row stride 160B = 16B-mult
    __shared__ __attribute__((aligned(16))) uint16_t sW3[FDIM * FDIM];
    __shared__ __attribute__((aligned(16))) uint16_t sW4[FDIM * FDIM];
    __shared__ __attribute__((aligned(16))) uint16_t sT[128 * TSTRIDE];

    const int tid   = threadIdx.x;
    const int wave  = tid >> 5;
    const int lane  = tid & 31;
    const int laneM = lane & 15;
    const int kgrp  = (lane >> 4) * 8;
    const long base = (long)blockIdx.x * 128;

    for (int i = tid; i < FDIM * FDIM / 2; i += 256) {
        ((uint32_t*)sW3)[i] = ((const uint32_t*)w3bf)[i];
        ((uint32_t*)sW4)[i] = ((const uint32_t*)w4bf)[i];
    }
    // load m tile (f32) -> bf16 LDS tile
    for (int i = tid; i < 128 * (FDIM / 2); i += 256) {
        const int r  = i >> 5;          // 32 uints per row
        const int c2 = i & 31;
        long rg = base + r;
        if (rg >= N) rg = N - 1;
        const float* p = m + rg * FDIM + c2 * 2;
        ((uint32_t*)sT)[r * (TSTRIDE / 2) + c2] = pk_bf16(p[0], p[1]);
    }
    __syncthreads();

    const v8f vzero = {};
    const uint16_t* arow = sT + (wave * 16 + laneM) * TSTRIDE;

    // GEMM3 with W3
    v8f acc[4];
#pragma unroll
    for (int t = 0; t < 4; ++t) acc[t] = vzero;
#pragma unroll
    for (int ks = 0; ks < 2; ++ks) {
        const int k0 = ks * 32;
        Frag16 afrag;
        afrag.q[0] = *(const uint4*)(arow + k0 + kgrp);
        afrag.q[1] = *(const uint4*)(arow + k0 + 16 + kgrp);
#pragma unroll
        for (int t = 0; t < 4; ++t) {
            const uint16_t* wrow = sW3 + (t * 16 + laneM) * FDIM + k0 + kgrp;
            Frag16 bfrag;
            bfrag.q[0] = *(const uint4*)(wrow);
            bfrag.q[1] = *(const uint4*)(wrow + 16);
            acc[t] = __builtin_amdgcn_wmma_f32_16x16x32_bf16(
                false, afrag.bf, false, bfrag.bf, (short)0, acc[t], false, false);
        }
    }
    // softplus + b3 -> overwrite own rows of sT (wave-private rows, no barrier)
#pragma unroll
    for (int t = 0; t < 4; ++t) {
        const int col  = t * 16 + laneM;
        const float bb = b3[col];
#pragma unroll
        for (int v = 0; v < 8; ++v) {
            const int rloc = wave * 16 + v + kgrp;
            sT[rloc * TSTRIDE + col] = f2bf_fast(sp_ln2(acc[t][v] + bb));
        }
    }

    // GEMM4 with W4
#pragma unroll
    for (int t = 0; t < 4; ++t) acc[t] = vzero;
#pragma unroll
    for (int ks = 0; ks < 2; ++ks) {
        const int k0 = ks * 32;
        Frag16 afrag;
        afrag.q[0] = *(const uint4*)(arow + k0 + kgrp);
        afrag.q[1] = *(const uint4*)(arow + k0 + 16 + kgrp);
#pragma unroll
        for (int t = 0; t < 4; ++t) {
            const uint16_t* wrow = sW4 + (t * 16 + laneM) * FDIM + k0 + kgrp;
            Frag16 bfrag;
            bfrag.q[0] = *(const uint4*)(wrow);
            bfrag.q[1] = *(const uint4*)(wrow + 16);
            acc[t] = __builtin_amdgcn_wmma_f32_16x16x32_bf16(
                false, afrag.bf, false, bfrag.bf, (short)0, acc[t], false, false);
        }
    }
    // softplus + b4 -> global out
#pragma unroll
    for (int t = 0; t < 4; ++t) {
        const int col  = t * 16 + laneM;
        const float bb = b4[col];
#pragma unroll
        for (int v = 0; v < 8; ++v) {
            const int  rloc = wave * 16 + v + kgrp;
            const long rg   = base + rloc;
            if (rg < N) out[rg * FDIM + col] = sp_ln2(acc[t][v] + bb);
        }
    }
}

// ---------------------------------------------------------------------------
extern "C" void kernel_launch(void* const* d_in, const int* in_sizes, int n_in,
                              void* d_out, int out_size, void* d_ws, size_t ws_size,
                              hipStream_t stream) {
    const float* bf     = (const float*)d_in[0];
    const float* hfeat  = (const float*)d_in[1];
    const float* h_edge = (const float*)d_in[2];
    const float* cutoff = (const float*)d_in[3];
    const int*   knn    = (const int*)d_in[4];
    const float* W1     = (const float*)d_in[5];
    const float* b1     = (const float*)d_in[6];
    const float* W2     = (const float*)d_in[7];
    const float* b2     = (const float*)d_in[8];
    const float* W3     = (const float*)d_in[9];
    const float* b3     = (const float*)d_in[10];
    const float* W4     = (const float*)d_in[11];
    const float* b4     = (const float*)d_in[12];

    const int N = in_sizes[1] / FDIM;   // h is [N, 64]

    // workspace layout: bf16 weights (96KB) then m [N,64] f32
    uint16_t* wbf = (uint16_t*)d_ws;
    float*    mws = (float*)((char*)d_ws + 49152 * sizeof(uint16_t));

    cvt_weights_kernel<<<192, 256, 0, stream>>>(W1, W2, W3, W4, wbf);

    const int blocksA = (N + 3) / 4;
    schnet_filter_kernel<<<blocksA, 256, 0, stream>>>(
        bf, hfeat, h_edge, cutoff, knn, b1, b2,
        wbf, wbf + 32768, mws, N);

    const int blocksB = (N + 127) / 128;
    schnet_out_kernel<<<blocksB, 256, 0, stream>>>(
        mws, b3, b4, wbf + 40960, wbf + 45056, (float*)d_out, N);
}
